// GIN_ZINC_v2_77008763617631
// MI455X (gfx1250) — compile-verified, matrix operand order checked by reading, not run on hardware
//
#include <hip/hip_runtime.h>
#include <hip/hip_bf16.h>

typedef __bf16 bf16_t;
typedef __attribute__((ext_vector_type(16))) __bf16 v16bf;
typedef __attribute__((ext_vector_type(8)))  __bf16 v8bf;
typedef __attribute__((ext_vector_type(8)))  float  v8f;
typedef unsigned int v4u  __attribute__((ext_vector_type(4)));
typedef int          v8i_ __attribute__((ext_vector_type(8)));
typedef int          v4i_ __attribute__((ext_vector_type(4)));

union frag16 { v16bf v; v8bf h2[2]; };

#define BN_EPS 1e-5f

__device__ inline void atomAddF(float* p, float v) {
#if defined(__AMDGCN__)
    unsafeAtomicAdd(p, v);   // global_atomic_add_f32
#else
    atomicAdd(p, v);
#endif
}

// ---------------------------------------------------------------------------
// TDM: async 2D tile load (128x128 bf16, contiguous) from global into LDS.
// D# per CDNA5 ISA ch.8: group0 {count=1, lds_addr, global_addr, type=2},
// group1 {data_size=2B, tensor 128x128, tile 128x128, dim0_stride=128}.
// This toolchain exposes the 6-arg builtin: (g0, g1, g2, g3, g4, cpol).
// Issued by one wave; completion via s_wait_tensorcnt.
// ---------------------------------------------------------------------------
__device__ inline void tdm_load_w128(const bf16_t* gsrc, bf16_t* lds_dst) {
#if defined(__AMDGCN__)
    unsigned long long ga = (unsigned long long)(size_t)gsrc;
    unsigned int       la = (unsigned int)(size_t)(void*)lds_dst;
    v4u g0;
    g0.x = 1u;                                            // count=1, user mode
    g0.y = la;                                            // lds_addr
    g0.z = (unsigned int)(ga & 0xffffffffu);              // global_addr[31:0]
    g0.w = (unsigned int)((ga >> 32) & 0x01ffffffu)       // global_addr[56:32]
         | (2u << 30);                                    // type=2 (image)
    v8i_ g1;
    g1[0] = (int)(1u << 16);          // workgroup_mask=0, data_size=1 (2 bytes)
    g1[1] = (int)(128u << 16);        // tensor_dim0[15:0] at bits [63:48]
    g1[2] = (int)(128u << 16);        // tensor_dim0 hi=0 | tensor_dim1 lo=128
    g1[3] = (int)(128u << 16);        // tensor_dim1 hi=0 | tile_dim0=128
    g1[4] = 128;                      // tile_dim1=128, tile_dim2=0
    g1[5] = 128;                      // tensor_dim0_stride[31:0] = 128 elements
    g1[6] = 0;
    g1[7] = 0;
    v4i_ z4 = {0, 0, 0, 0};
    v8i_ z8 = {0, 0, 0, 0, 0, 0, 0, 0};
    __builtin_amdgcn_tensor_load_to_lds(g0, g1, z4, z4, z8, 0);
#else
    (void)gsrc; (void)lds_dst;
#endif
}

// ---------------------------------------------------------------------------
// Prep: transpose-convert a [128(k),128(n)] f32 weight to bf16 [n][k]
// ---------------------------------------------------------------------------
__global__ void wt_cvt_kernel(const float* __restrict__ W, bf16_t* __restrict__ out) {
    int n = blockIdx.x, k = threadIdx.x;      // 128 x 128
    out[n * 128 + k] = (bf16_t)W[k * 128 + n];
}

// Fold BN (eval mode) into per-channel scale/shift
__global__ void bn_fold_kernel(const float* __restrict__ gam, const float* __restrict__ bet,
                               const float* __restrict__ mea, const float* __restrict__ var,
                               float* __restrict__ scale, float* __restrict__ shift) {
    int i = threadIdx.x;                      // 128
    float sc = gam[i] * rsqrtf(var[i] + BN_EPS);
    scale[i] = sc;
    shift[i] = bet[i] - mea[i] * sc;
}

// ---------------------------------------------------------------------------
// Node init: h0 = concat(atom_emb[x], BN(relu(pe@w1+b1)@w2+b2)) @ in_w + in_b
// ---------------------------------------------------------------------------
__global__ __launch_bounds__(128)
void node_init_kernel(float* __restrict__ hout,
                      const int* __restrict__ x, const float* __restrict__ pe,
                      const float* __restrict__ atom_emb,
                      const float* __restrict__ pw1, const float* __restrict__ pb1,
                      const float* __restrict__ pw2, const float* __restrict__ pb2,
                      const float* __restrict__ bng, const float* __restrict__ bnb,
                      const float* __restrict__ bnm, const float* __restrict__ bnv,
                      const float* __restrict__ in_w, const float* __restrict__ in_b,
                      int Nn)
{
    __shared__ float sIn[16][80];
    __shared__ float sHid[16][16];
    const int tid  = threadIdx.x;
    const int base = blockIdx.x * 16;

    for (int it = 0; it < 2; ++it) {
        int task = tid + it * 128;
        int n = task >> 4, j = task & 15;
        int gn = base + n;
        float s = pb1[j];
        if (gn < Nn) {
            #pragma unroll
            for (int k = 0; k < 20; ++k) s += pe[(size_t)gn * 20 + k] * pw1[k * 16 + j];
        }
        sHid[n][j] = s > 0.f ? s : 0.f;
    }
    __syncthreads();

    for (int it = 0; it < 2; ++it) {
        int task = tid + it * 128;
        int n = task >> 4, j = task & 15;
        float z = pb2[j];
        #pragma unroll
        for (int k = 0; k < 16; ++k) z += sHid[n][k] * pw2[k * 16 + j];
        z = (z - bnm[j]) * rsqrtf(bnv[j] + BN_EPS) * bng[j] + bnb[j];
        sIn[n][64 + j] = z;
    }
    for (int it = 0; it < 8; ++it) {
        int task = tid + it * 128;
        int n = task >> 6, c = task & 63;
        int gn = base + n;
        int a = (gn < Nn) ? x[gn] : 0;
        sIn[n][c] = atom_emb[a * 64 + c];
    }
    __syncthreads();

    const int j = tid;
    for (int n = 0; n < 16; ++n) {
        int gn = base + n;
        if (gn >= Nn) break;
        float s = in_b[j];
        #pragma unroll 8
        for (int k = 0; k < 80; ++k) s += sIn[n][k] * in_w[k * 128 + j];
        hout[(size_t)gn * 128 + j] = s;
    }
}

// ---------------------------------------------------------------------------
// Utility kernels
// ---------------------------------------------------------------------------
__global__ void copy_f4_kernel(float4* __restrict__ dst, const float4* __restrict__ src, size_t n4) {
    size_t i = (size_t)blockIdx.x * blockDim.x + threadIdx.x;
    size_t stride = (size_t)gridDim.x * blockDim.x;
    for (; i < n4; i += stride) dst[i] = src[i];
}

__global__ void zero_f_kernel(float* __restrict__ p, size_t n) {
    size_t i = (size_t)blockIdx.x * blockDim.x + threadIdx.x;
    size_t stride = (size_t)gridDim.x * blockDim.x;
    for (; i < n; i += stride) p[i] = 0.f;
}

__global__ __launch_bounds__(256)
void scatter_edges_kernel(float* __restrict__ t, const float* __restrict__ h,
                          const int* __restrict__ ei, int Ee) {
    int warp = (int)(((size_t)blockIdx.x * blockDim.x + threadIdx.x) >> 5);
    int lane = threadIdx.x & 31;
    if (warp >= Ee) return;
    int src = ei[warp];
    int dst = ei[Ee + warp];
    const float* hs = h + (size_t)src * 128;
    float*       td = t + (size_t)dst * 128;
    #pragma unroll
    for (int i = 0; i < 4; ++i) atomAddF(&td[lane + 32 * i], hs[lane + 32 * i]);
}

__global__ __launch_bounds__(256)
void pool_nodes_kernel(float* __restrict__ g, const float* __restrict__ h,
                       const int* __restrict__ batch, int Nn) {
    int warp = (int)(((size_t)blockIdx.x * blockDim.x + threadIdx.x) >> 5);
    int lane = threadIdx.x & 31;
    if (warp >= Nn) return;
    int b = batch[warp];
    const float* hs = h + (size_t)warp * 128;
    float*       gd = g + (size_t)b * 128;
    #pragma unroll
    for (int i = 0; i < 4; ++i) atomAddF(&gd[lane + 32 * i], hs[lane + 32 * i]);
}

// ---------------------------------------------------------------------------
// Fused GIN layer: out = relu(BN(relu(in@W1+b1)@W2+b2))
// 256 threads = 8 waves; each wave owns 16 rows. A tile staged bf16 in LDS;
// W1/W2 fetched by the Tensor Data Mover; GEMM1 result (bf16) overwrites the
// A tile in-place (per-wave private rows -> only intra-wave DS ordering).
// ---------------------------------------------------------------------------
__global__ __launch_bounds__(256)
void gin_layer_wmma_kernel(const float* __restrict__ in, float* __restrict__ out,
                           const bf16_t* __restrict__ W1t, const float* __restrict__ b1,
                           const bf16_t* __restrict__ W2t, const float* __restrict__ b2,
                           const float* __restrict__ scale, const float* __restrict__ shift,
                           int M)
{
    __shared__ __align__(16) bf16_t sA[128 * 128];   // A tile, then U tile (32 KB)
    __shared__ __align__(16) bf16_t sW1[128 * 128];  // 32 KB
    __shared__ __align__(16) bf16_t sW2[128 * 128];  // 32 KB

    const int tid      = threadIdx.x;
    const int lane     = tid & 31;
    const int wave     = tid >> 5;
    const int tileBase = blockIdx.x * 128;
    const bool full    = (tileBase + 128 <= M);

    // --- Weights via TDM (wave 0 issues, waits TENSORcnt, then block barrier)
    if (tid < 32) {
        tdm_load_w128(W1t, sW1);
        tdm_load_w128(W2t, sW2);
        __builtin_amdgcn_s_wait_tensorcnt(0);
    }

    // --- Stage A tile: 128x128 f32 -> bf16 LDS ---
    if (full) {
        #pragma unroll
        for (int i = 0; i < 16; ++i) {
            int f4  = tid + i * 256;
            int row = f4 >> 5;
            int c4  = (f4 & 31) * 4;
            float4 v = *(const float4*)(in + (size_t)(tileBase + row) * 128 + c4);
            bf16_t* p = &sA[row * 128 + c4];
            p[0] = (bf16_t)v.x; p[1] = (bf16_t)v.y; p[2] = (bf16_t)v.z; p[3] = (bf16_t)v.w;
        }
    } else {
        #pragma unroll
        for (int i = 0; i < 16; ++i) {
            int f4  = tid + i * 256;
            int row = f4 >> 5;
            int c4  = (f4 & 31) * 4;
            int gr  = tileBase + row;
            float4 v = make_float4(0.f, 0.f, 0.f, 0.f);
            if (gr < M) v = *(const float4*)(in + (size_t)gr * 128 + c4);
            bf16_t* p = &sA[row * 128 + c4];
            p[0] = (bf16_t)v.x; p[1] = (bf16_t)v.y; p[2] = (bf16_t)v.z; p[3] = (bf16_t)v.w;
        }
    }
    __syncthreads();

    const int m0   = wave * 16;
    const int hh   = lane >> 4;
    const int lrow = lane & 15;

    // Per-lane epilogue constants (col = nt*16 + lrow)
    float bb1[8], bb2[8], ssc[8], ssh[8];
    #pragma unroll
    for (int nt = 0; nt < 8; ++nt) {
        int col = nt * 16 + lrow;
        bb1[nt] = b1[col];
        bb2[nt] = b2[col];
        ssc[nt] = scale[col];
        ssh[nt] = shift[col];
    }

    // ---- GEMM1: U = relu(A @ W1 + b1), written bf16 back into sA ----
    frag16 a[4];
    #pragma unroll
    for (int kb = 0; kb < 4; ++kb) {
        const bf16_t* pr = &sA[(m0 + lrow) * 128 + kb * 32 + hh * 8];
        a[kb].h2[0] = *(const v8bf*)(pr);
        a[kb].h2[1] = *(const v8bf*)(pr + 16);
    }
    #pragma unroll
    for (int nt = 0; nt < 8; ++nt) {
        v8f acc = {0.f, 0.f, 0.f, 0.f, 0.f, 0.f, 0.f, 0.f};
        const int n = nt * 16 + lrow;
        #pragma unroll
        for (int kb = 0; kb < 4; ++kb) {
            frag16 b;
            const bf16_t* pc = &sW1[n * 128 + kb * 32 + hh * 8];
            b.h2[0] = *(const v8bf*)(pc);
            b.h2[1] = *(const v8bf*)(pc + 16);
            acc = __builtin_amdgcn_wmma_f32_16x16x32_bf16(
                false, a[kb].v, false, b.v, (short)0, acc, false, false);
        }
        // D: vgpr r, lane l -> row m0 + r + 8*(l/16), col n. Rows are this
        // wave's own rows, so in-wave DS ordering suffices (no barrier).
        #pragma unroll
        for (int r = 0; r < 8; ++r) {
            float v = acc[r] + bb1[nt];
            v = v > 0.f ? v : 0.f;
            sA[(m0 + r + hh * 8) * 128 + n] = (bf16_t)v;
        }
    }

    // ---- GEMM2: out = relu((U @ W2 + b2)*scale + shift) ----
    frag16 a2[4];
    #pragma unroll
    for (int kb = 0; kb < 4; ++kb) {
        const bf16_t* pr = &sA[(m0 + lrow) * 128 + kb * 32 + hh * 8];
        a2[kb].h2[0] = *(const v8bf*)(pr);
        a2[kb].h2[1] = *(const v8bf*)(pr + 16);
    }
    #pragma unroll
    for (int nt = 0; nt < 8; ++nt) {
        v8f acc = {0.f, 0.f, 0.f, 0.f, 0.f, 0.f, 0.f, 0.f};
        const int n = nt * 16 + lrow;
        #pragma unroll
        for (int kb = 0; kb < 4; ++kb) {
            frag16 b;
            const bf16_t* pc = &sW2[n * 128 + kb * 32 + hh * 8];
            b.h2[0] = *(const v8bf*)(pc);
            b.h2[1] = *(const v8bf*)(pc + 16);
            acc = __builtin_amdgcn_wmma_f32_16x16x32_bf16(
                false, a2[kb].v, false, b.v, (short)0, acc, false, false);
        }
        if (full) {
            #pragma unroll
            for (int r = 0; r < 8; ++r) {
                int grow = tileBase + m0 + r + hh * 8;
                float v = (acc[r] + bb2[nt]) * ssc[nt] + ssh[nt];
                v = v > 0.f ? v : 0.f;
                out[(size_t)grow * 128 + n] = v;
            }
        } else {
            #pragma unroll
            for (int r = 0; r < 8; ++r) {
                int grow = tileBase + m0 + r + hh * 8;
                if (grow < M) {
                    float v = (acc[r] + bb2[nt]) * ssc[nt] + ssh[nt];
                    v = v > 0.f ? v : 0.f;
                    out[(size_t)grow * 128 + n] = v;
                }
            }
        }
    }
}

// ---------------------------------------------------------------------------
// Single WMMA GEMM (readout): out = relu(in@W + bias)
// ---------------------------------------------------------------------------
__global__ __launch_bounds__(256)
void gemm128_wmma_kernel(const float* __restrict__ in, float* __restrict__ out,
                         const bf16_t* __restrict__ Wt, const float* __restrict__ bias,
                         int M)
{
    __shared__ __align__(16) bf16_t sA[128 * 128];
    __shared__ __align__(16) bf16_t sW[128 * 128];

    const int tid      = threadIdx.x;
    const int lane     = tid & 31;
    const int wave     = tid >> 5;
    const int tileBase = blockIdx.x * 128;
    const bool full    = (tileBase + 128 <= M);

    if (tid < 32) {
        tdm_load_w128(Wt, sW);
        __builtin_amdgcn_s_wait_tensorcnt(0);
    }
    #pragma unroll
    for (int i = 0; i < 16; ++i) {
        int f4  = tid + i * 256;
        int row = f4 >> 5;
        int c4  = (f4 & 31) * 4;
        int gr  = tileBase + row;
        float4 v = make_float4(0.f, 0.f, 0.f, 0.f);
        if (full || gr < M) v = *(const float4*)(in + (size_t)gr * 128 + c4);
        bf16_t* p = &sA[row * 128 + c4];
        p[0] = (bf16_t)v.x; p[1] = (bf16_t)v.y; p[2] = (bf16_t)v.z; p[3] = (bf16_t)v.w;
    }
    __syncthreads();

    const int m0   = wave * 16;
    const int hh   = lane >> 4;
    const int lrow = lane & 15;

    float bb[8];
    #pragma unroll
    for (int nt = 0; nt < 8; ++nt) bb[nt] = bias[nt * 16 + lrow];

    frag16 a[4];
    #pragma unroll
    for (int kb = 0; kb < 4; ++kb) {
        const bf16_t* pr = &sA[(m0 + lrow) * 128 + kb * 32 + hh * 8];
        a[kb].h2[0] = *(const v8bf*)(pr);
        a[kb].h2[1] = *(const v8bf*)(pr + 16);
    }
    #pragma unroll
    for (int nt = 0; nt < 8; ++nt) {
        v8f acc = {0.f, 0.f, 0.f, 0.f, 0.f, 0.f, 0.f, 0.f};
        const int n = nt * 16 + lrow;
        #pragma unroll
        for (int kb = 0; kb < 4; ++kb) {
            frag16 b;
            const bf16_t* pc = &sW[n * 128 + kb * 32 + hh * 8];
            b.h2[0] = *(const v8bf*)(pc);
            b.h2[1] = *(const v8bf*)(pc + 16);
            acc = __builtin_amdgcn_wmma_f32_16x16x32_bf16(
                false, a[kb].v, false, b.v, (short)0, acc, false, false);
        }
        #pragma unroll
        for (int r = 0; r < 8; ++r) {
            int grow = tileBase + m0 + r + hh * 8;
            if (full || grow < M) {
                float v = acc[r] + bb[nt];
                v = v > 0.f ? v : 0.f;
                out[(size_t)grow * 128 + n] = v;
            }
        }
    }
}

// ---------------------------------------------------------------------------
// Final readout: out[g] = gr[g] . r_w2 + r_b2
// ---------------------------------------------------------------------------
__global__ void readout_final_kernel(float* __restrict__ out, const float* __restrict__ gr,
                                     const float* __restrict__ w2, const float* __restrict__ b2,
                                     int Gn) {
    int g = blockIdx.x * blockDim.x + threadIdx.x;
    if (g >= Gn) return;
    float s = b2[0];
    #pragma unroll 8
    for (int c = 0; c < 128; ++c) s += gr[(size_t)g * 128 + c] * w2[c];
    out[g] = s;
}

// ---------------------------------------------------------------------------
// Host launcher
// ---------------------------------------------------------------------------
extern "C" void kernel_launch(void* const* d_in, const int* in_sizes, int n_in,
                              void* d_out, int out_size, void* d_ws, size_t ws_size,
                              hipStream_t stream) {
    // 0:x 1:pe 2:edge_index 3:batch
    // 4:atom_emb 5:pe_w1 6:pe_b1 7:pe_w2 8:pe_b2 9..12:pe_bn(g,b,m,v) 13:in_w 14:in_b
    // 15+8l: layer l -> w1,b1,w2,b2,bn.gamma,bn.beta,bn.mean,bn.var
    // 47:r_w1 48:r_b1 49:r_w2 50:r_b2
    const int Nn = in_sizes[0];
    const int Ee = in_sizes[2] / 2;
    const int Gn = out_size;

    const int*   x      = (const int*)d_in[0];
    const float* pe     = (const float*)d_in[1];
    const int*   ei     = (const int*)d_in[2];
    const int*   batch  = (const int*)d_in[3];
    const float* a_emb  = (const float*)d_in[4];
    const float* pw1    = (const float*)d_in[5];
    const float* pb1    = (const float*)d_in[6];
    const float* pw2    = (const float*)d_in[7];
    const float* pb2    = (const float*)d_in[8];
    const float* pbng   = (const float*)d_in[9];
    const float* pbnb   = (const float*)d_in[10];
    const float* pbnm   = (const float*)d_in[11];
    const float* pbnv   = (const float*)d_in[12];
    const float* in_w   = (const float*)d_in[13];
    const float* in_b   = (const float*)d_in[14];
    const float* r_w1   = (const float*)d_in[47];
    const float* r_b1   = (const float*)d_in[48];
    const float* r_w2   = (const float*)d_in[49];
    const float* r_b2   = (const float*)d_in[50];

    char* ws = (char*)d_ws;
    const size_t SZH = (size_t)Nn * 128 * sizeof(float);
    float*  buf0  = (float*)ws;                                    // h
    float*  buf1  = (float*)(ws + SZH);                            // t scratch
    float*  gacc  = (float*)(ws + 2 * SZH);                        // [G,128]
    float*  gr    = (float*)(ws + 2 * SZH + (size_t)Gn * 128 * 4); // [G,128]
    bf16_t* wts   = (bf16_t*)(ws + 2 * SZH + 2 * (size_t)Gn * 128 * 4);
    float*  scale = (float*)((char*)wts + (size_t)9 * 128 * 128 * sizeof(bf16_t));
    float*  shift = scale + 4 * 128;

    // Prep: transpose/convert weights, fold BN
    for (int l = 0; l < 4; ++l) {
        const float* w1 = (const float*)d_in[15 + 8 * l + 0];
        const float* w2 = (const float*)d_in[15 + 8 * l + 2];
        wt_cvt_kernel<<<128, 128, 0, stream>>>(w1, wts + (size_t)(2 * l) * 128 * 128);
        wt_cvt_kernel<<<128, 128, 0, stream>>>(w2, wts + (size_t)(2 * l + 1) * 128 * 128);
        bn_fold_kernel<<<1, 128, 0, stream>>>((const float*)d_in[15 + 8 * l + 4],
                                              (const float*)d_in[15 + 8 * l + 5],
                                              (const float*)d_in[15 + 8 * l + 6],
                                              (const float*)d_in[15 + 8 * l + 7],
                                              scale + l * 128, shift + l * 128);
    }
    wt_cvt_kernel<<<128, 128, 0, stream>>>(r_w1, wts + (size_t)8 * 128 * 128);

    // Node init -> h
    node_init_kernel<<<(Nn + 15) / 16, 128, 0, stream>>>(
        buf0, x, pe, a_emb, pw1, pb1, pw2, pb2, pbng, pbnb, pbnm, pbnv, in_w, in_b, Nn);

    // GIN layers (h stays in buf0; buf1 holds t = h + neighbor sum)
    const int gemmGrid = (Nn + 127) / 128;
    for (int l = 0; l < 4; ++l) {
        copy_f4_kernel<<<4096, 256, 0, stream>>>((float4*)buf1, (const float4*)buf0,
                                                 (size_t)Nn * 128 / 4);
        scatter_edges_kernel<<<(Ee + 7) / 8, 256, 0, stream>>>(buf1, buf0, ei, Ee);
        gin_layer_wmma_kernel<<<gemmGrid, 256, 0, stream>>>(
            buf1, buf0,
            wts + (size_t)(2 * l) * 128 * 128, (const float*)d_in[15 + 8 * l + 1],
            wts + (size_t)(2 * l + 1) * 128 * 128, (const float*)d_in[15 + 8 * l + 3],
            scale + l * 128, shift + l * 128, Nn);
    }

    // Pool + readout
    zero_f_kernel<<<1024, 256, 0, stream>>>(gacc, (size_t)2 * Gn * 128);
    pool_nodes_kernel<<<(Nn + 7) / 8, 256, 0, stream>>>(gacc, buf0, batch, Nn);
    gemm128_wmma_kernel<<<(Gn + 127) / 128, 256, 0, stream>>>(
        gacc, gr, wts + (size_t)8 * 128 * 128, r_b1, Gn);
    readout_final_kernel<<<(Gn + 255) / 256, 256, 0, stream>>>((float*)d_out, gr, r_w2, r_b2, Gn);
}
